// LTCEncoder_78847009620178
// MI455X (gfx1250) — compile-verified, matrix operand order checked by reading, not run on hardware
//
#include <hip/hip_runtime.h>
#include <math.h>

// LTC encoder: 2-layer liquid-time-constant RNN, T=1024 sequential steps.
// Latency-bound recurrence -> weights resident in VGPRs as WMMA B-tiles,
// state h0/h1 in registers (f32, C-layout) + LDS (f16, A-layout source),
// v_wmma_f32_16x16x32_f16 for all matmuls, LN via shfl + LDS f32 atomics.
// Elementwise math is branch-free hardware-TRANS (v_exp/v_log/v_rcp/v_rsq).

typedef __attribute__((ext_vector_type(16))) _Float16 v16h;
typedef __attribute__((ext_vector_type(8)))  _Float16 v8h;
typedef __attribute__((ext_vector_type(8)))  float    v8f;
typedef __attribute__((ext_vector_type(4)))  float    v4f;

#define BB   256
#define TT   1024
#define FF   64
#define HH   128
#define LNE  1e-5f
#define HPAD (HH + 8)   // 136 halves per row: 272B stride -> bank-conflict free
#define FPAD (FF + 8)

// CDNA5 16-bit A/B operand half-index -> K mapping (within a K=32 unit):
// lane<16: i<8 -> K=i,     i>=8 -> K=i+8
// lane>=16: i<8 -> K=i+8,  i>=8 -> K=i+16
__device__ __forceinline__ int kmap(int i, int laneHi) {
    return i + (i >= 8 ? 8 : 0) + (laneHi ? 8 : 0);
}

// Build one v16h A-operand unit (K base = 32*u) from a row-major f16 row in LDS.
// Halves 0..7 and 8..15 are each contiguous 16-byte chunks -> two ds_load_b128.
__device__ __forceinline__ v16h load_aunit(const _Float16* __restrict__ row, int u, int hi) {
    const v8h lo = *(const v8h*)(row + 32 * u + 8 * hi);
    const v8h hh = *(const v8h*)(row + 32 * u + 8 * hi + 16);
    v16h a;
#pragma unroll
    for (int i = 0; i < 8; ++i) { a[i] = lo[i]; a[i + 8] = hh[i]; }
    return a;
}

// Gather a WMMA B-tile (K x 16) for output columns [n0, n0+16) of a row-major
// weight W[N][K] (we need B[k][n] = W[n0+n][k]). One-time cost at kernel start.
template <int KD, int NU>
__device__ __forceinline__ void load_btiles(const float* __restrict__ W, int n, int hi,
                                            v16h* __restrict__ dst) {
#pragma unroll
    for (int u = 0; u < NU; ++u) {
        v16h b;
#pragma unroll
        for (int i = 0; i < 16; ++i)
            b[i] = (_Float16)W[(size_t)n * KD + 32 * u + kmap(i, hi)];
        dst[u] = b;
    }
}

// Branch-free transcendentals on the serial critical path.
__device__ __forceinline__ float fast_tanh(float x) {
    float a = fabsf(x);
    float e = __expf(2.0f * a);                        // v_exp_f32; inf-safe
    float t = 1.0f - 2.0f * __builtin_amdgcn_rcpf(e + 1.0f);
    return copysignf(t, x);
}
__device__ __forceinline__ float fast_softplus(float z) {
    // max(z,0) + log(1 + exp(-|z|)): no overflow, no branches
    return fmaxf(z, 0.0f) + __logf(1.0f + __expf(-fabsf(z)));
}

#define WMMA(A, Bm, C) \
    __builtin_amdgcn_wmma_f32_16x16x32_f16(false, (A), false, (Bm), (short)0, (C), false, false)

__global__ __launch_bounds__(256, 1)
void ltc_encoder_kernel(const float* __restrict__ x,
                        const float* __restrict__ Wh0, const float* __restrict__ bh0,
                        const float* __restrict__ Wx0, const float* __restrict__ bx0,
                        const float* __restrict__ Wt0, const float* __restrict__ bt0,
                        const float* __restrict__ tau0, const float* __restrict__ g0,
                        const float* __restrict__ be0,
                        const float* __restrict__ Wh1, const float* __restrict__ bh1,
                        const float* __restrict__ Wx1, const float* __restrict__ bx1,
                        const float* __restrict__ Wt1, const float* __restrict__ bt1,
                        const float* __restrict__ tau1, const float* __restrict__ g1,
                        const float* __restrict__ be1,
                        float* __restrict__ out) {
    __shared__ _Float16 h0A[16][HPAD];   // f16 state, A-operand source (row-major [m][k])
    __shared__ _Float16 h1A[16][HPAD];
    __shared__ _Float16 xA[16][FPAD];    // staged x_t tile
    __shared__ float    red[2][2][16];   // [layer][sum|sumsq][row]

    const int tid  = threadIdx.x;
    const int lane = tid & 31;
    const int wave = tid >> 5;      // 0..7 -> N-tile
    const int grp  = lane >> 4;     // 0/1: half-wave (K-half for A/B, M-offset for C)
    const int nl   = lane & 15;
    const int n    = wave * 16 + nl;            // output column owned by this lane
    const int b0   = blockIdx.x * 16;           // batch tile base

    // ---- resident weight B-tiles (160 VGPRs) ----
    v16h wh0r[4], wx0r[2], wt0r[2], wh1r[4], wx1r[4], wt1r[4];
    load_btiles<HH, 4>(Wh0, n, grp, wh0r);
    load_btiles<FF, 2>(Wx0, n, grp, wx0r);
    load_btiles<FF, 2>(Wt0, n, grp, wt0r);
    load_btiles<HH, 4>(Wh1, n, grp, wh1r);
    load_btiles<HH, 4>(Wx1, n, grp, wx1r);
    load_btiles<HH, 4>(Wt1, n, grp, wt1r);

    // per-column parameters
    const float b0v = bh0[n] + bx0[n], bt0v = bt0[n], tb0 = tau0[n], g0v = g0[n], e0v = be0[n];
    const float b1v = bh1[n] + bx1[n], bt1v = bt1[n], tb1 = tau1[n], g1v = g1[n], e1v = be1[n];

    // h state in registers (C layout: VGPR r -> row m = r + 8*grp, col n)
    float h0c[8], h1c[8];
#pragma unroll
    for (int r = 0; r < 8; ++r) { h0c[r] = 0.f; h1c[r] = 0.f; }
    for (int i = tid; i < 16 * HPAD; i += 256) {
        (&h0A[0][0])[i] = (_Float16)0.f;
        (&h1A[0][0])[i] = (_Float16)0.f;
    }
    __syncthreads();

    const float invH = 1.0f / (float)HH;

    for (int t = 0; t < TT; ++t) {
        // ---- stage x_t (coalesced float4) + zero LN accumulators ----
        {
            const int m  = tid >> 4;
            const int f0 = (tid & 15) * 4;
            const v4f xv = *(const v4f*)(x + ((size_t)(b0 + m) * TT + t) * FF + f0);
#pragma unroll
            for (int j = 0; j < 4; ++j) xA[m][f0 + j] = (_Float16)xv[j];
        }
        if (tid < 64) (&red[0][0][0])[tid] = 0.f;
        __syncthreads();

        // ================= layer 0 =================
        v8f accF, accT;
#pragma unroll
        for (int r = 0; r < 8; ++r) { accF[r] = b0v; accT[r] = bt0v; }
#pragma unroll
        for (int u = 0; u < 4; ++u) {
            v16h a = load_aunit(&h0A[nl][0], u, grp);
            accF = WMMA(a, wh0r[u], accF);
        }
#pragma unroll
        for (int u = 0; u < 2; ++u) {
            v16h ax = load_aunit(&xA[nl][0], u, grp);
            accF = WMMA(ax, wx0r[u], accF);
            accT = WMMA(ax, wt0r[u], accT);
        }
        float hn[8];
#pragma unroll
        for (int r = 0; r < 8; ++r) {
            float f   = fast_tanh(accF[r]);
            float tau = tb0 + fast_softplus(accT[r]);
            hn[r] = h0c[r] + (f - h0c[r]) * __builtin_amdgcn_rcpf(tau);   // DT = 1
        }
#pragma unroll
        for (int r = 0; r < 8; ++r) {
            float s = hn[r], q = hn[r] * hn[r];
#pragma unroll
            for (int msk = 1; msk < 16; msk <<= 1) {
                s += __shfl_xor(s, msk, 32);
                q += __shfl_xor(q, msk, 32);
            }
            if (nl == 0) {
                atomicAdd(&red[0][0][r + 8 * grp], s);
                atomicAdd(&red[0][1][r + 8 * grp], q);
            }
        }
        __syncthreads();
#pragma unroll
        for (int r = 0; r < 8; ++r) {
            const int m = r + 8 * grp;
            float mu  = red[0][0][m] * invH;
            float var = red[0][1][m] * invH - mu * mu;
            float v   = (hn[r] - mu) * __builtin_amdgcn_rsqf(var + LNE) * g0v + e0v;
            h0c[r]    = v;
            h0A[m][n] = (_Float16)v;
        }
        __syncthreads();

        // ================= layer 1 (input = h0) =================
#pragma unroll
        for (int r = 0; r < 8; ++r) { accF[r] = b1v; accT[r] = bt1v; }
#pragma unroll
        for (int u = 0; u < 4; ++u) {
            v16h a1 = load_aunit(&h1A[nl][0], u, grp);
            accF = WMMA(a1, wh1r[u], accF);
        }
#pragma unroll
        for (int u = 0; u < 4; ++u) {
            v16h a0 = load_aunit(&h0A[nl][0], u, grp);
            accF = WMMA(a0, wx1r[u], accF);
            accT = WMMA(a0, wt1r[u], accT);
        }
#pragma unroll
        for (int r = 0; r < 8; ++r) {
            float f   = fast_tanh(accF[r]);
            float tau = tb1 + fast_softplus(accT[r]);
            hn[r] = h1c[r] + (f - h1c[r]) * __builtin_amdgcn_rcpf(tau);
        }
#pragma unroll
        for (int r = 0; r < 8; ++r) {
            float s = hn[r], q = hn[r] * hn[r];
#pragma unroll
            for (int msk = 1; msk < 16; msk <<= 1) {
                s += __shfl_xor(s, msk, 32);
                q += __shfl_xor(q, msk, 32);
            }
            if (nl == 0) {
                atomicAdd(&red[1][0][r + 8 * grp], s);
                atomicAdd(&red[1][1][r + 8 * grp], q);
            }
        }
        __syncthreads();
#pragma unroll
        for (int r = 0; r < 8; ++r) {
            const int m = r + 8 * grp;
            float mu  = red[1][0][m] * invH;
            float var = red[1][1][m] * invH - mu * mu;
            float v   = (hn[r] - mu) * __builtin_amdgcn_rsqf(var + LNE) * g1v + e1v;
            h1c[r]    = v;
            h1A[m][n] = (_Float16)v;
        }
        __syncthreads();
    }

    // final h1 -> out (B,H) f32
#pragma unroll
    for (int r = 0; r < 8; ++r)
        out[(size_t)(b0 + r + 8 * grp) * HH + n] = h1c[r];
}

extern "C" void kernel_launch(void* const* d_in, const int* in_sizes, int n_in,
                              void* d_out, int out_size, void* d_ws, size_t ws_size,
                              hipStream_t stream) {
    (void)in_sizes; (void)n_in; (void)out_size; (void)d_ws; (void)ws_size;
    const float* x   = (const float*)d_in[0];
    const float* Wh0 = (const float*)d_in[1];
    const float* bh0 = (const float*)d_in[2];
    const float* Wx0 = (const float*)d_in[3];
    const float* bx0 = (const float*)d_in[4];
    const float* Wt0 = (const float*)d_in[5];
    const float* bt0 = (const float*)d_in[6];
    const float* ta0 = (const float*)d_in[7];
    const float* g0  = (const float*)d_in[8];
    const float* be0 = (const float*)d_in[9];
    const float* Wh1 = (const float*)d_in[10];
    const float* bh1 = (const float*)d_in[11];
    const float* Wx1 = (const float*)d_in[12];
    const float* bx1 = (const float*)d_in[13];
    const float* Wt1 = (const float*)d_in[14];
    const float* bt1 = (const float*)d_in[15];
    const float* ta1 = (const float*)d_in[16];
    const float* g1  = (const float*)d_in[17];
    const float* be1 = (const float*)d_in[18];

    dim3 grid(BB / 16), block(256);
    ltc_encoder_kernel<<<grid, block, 0, stream>>>(
        x, Wh0, bh0, Wx0, bx0, Wt0, bt0, ta0, g0, be0,
        Wh1, bh1, Wx1, bx1, Wt1, bt1, ta1, g1, be1,
        (float*)d_out);
}
